// att_80487687127253
// MI455X (gfx1250) — matrix operand order repaired and checked by reading, not compiled, not run on hardware
//
#include <hip/hip_runtime.h>
#include <hip/hip_bf16.h>

// Incremental attention, B=32, S=512, D=512.
// One persistent 1024-thread workgroup per batch (batches are independent);
// growing memory kept as bf16 in d_ws (32 MB, L2-resident on MI455X's 192 MB L2).
// Score GEMV runs on v_wmma_f32_16x16x32_bf16; softmax f32 (deterministic);
// weighted sum is a coalesced b128 VALU GEMV with f32 accumulation; f32 output.
// Per-step barrier count minimized to 6 (the 512-step recurrence is latency-bound).

#define Bsz 32
#define Ssz 512
#define Dsz 512
#define M2  1024   // 2*S rows of memory per batch

typedef __attribute__((ext_vector_type(16))) __bf16 v16bf;
typedef __attribute__((ext_vector_type(8)))  float  v8f;

union ABPack { v16bf v; uint4 q[2]; };

static __device__ __forceinline__ unsigned short f2b(float f) {
  // round-to-nearest-even f32 -> bf16
  unsigned u = __float_as_uint(f);
  unsigned r = u + 0x7FFFu + ((u >> 16) & 1u);
  return (unsigned short)(r >> 16);
}
static __device__ __forceinline__ float b2f(unsigned short h) {
  return __uint_as_float(((unsigned)h) << 16);
}
// monotonic float -> uint key (order-preserving), for deterministic atomic max
static __device__ __forceinline__ unsigned fkey(float f) {
  unsigned u = __float_as_uint(f);
  return (u & 0x80000000u) ? ~u : (u | 0x80000000u);
}
static __device__ __forceinline__ float funkey(unsigned k) {
  unsigned u = (k & 0x80000000u) ? (k ^ 0x80000000u) : ~k;
  return __uint_as_float(u);
}

__global__ __launch_bounds__(1024, 1)
void att_incremental_kernel(const float* __restrict__ enc,
                            const float* __restrict__ dec,
                            float* __restrict__ out,
                            unsigned short* __restrict__ memb_all)
{
  const int b    = blockIdx.x;
  const int tid  = threadIdx.x;
  const int lane = tid & 31;
  const int wave = tid >> 5;

  unsigned short* memb  = memb_all + (size_t)b * M2 * Dsz;
  const float*    encb  = enc + (size_t)b * Ssz * Dsz;
  const float*    decbg = dec + (size_t)b * Ssz * Dsz;
  float*          outb  = out + (size_t)b * Ssz * Dsz;

  __shared__ __align__(16) unsigned short s_decb[2][Dsz]; // query bf16, ping-pong
  __shared__ float    s_decf[2][Dsz];                     // query f32, ping-pong
  __shared__ float    s_sc[M2];                           // scores -> attention weights
  __shared__ float    s_redsum[32];                       // per-wave exp sums
  __shared__ unsigned s_mxkey;                            // keyed max (ds_max_u32)
  __shared__ float    s_part[16][Dsz];                    // weighted-sum partials (32 KB)

  // ---- init: enc (f32) -> bf16 memory rows [0,512); stage dec[:,0]; seed state ----
  for (int idx = tid * 4; idx < Ssz * Dsz; idx += 1024 * 4) {
    float4 f = *(const float4*)(encb + idx);
    uint2 p;
    p.x = (unsigned)f2b(f.x) | ((unsigned)f2b(f.y) << 16);
    p.y = (unsigned)f2b(f.z) | ((unsigned)f2b(f.w) << 16);
    *(uint2*)(memb + idx) = p;
  }
  if (tid < Dsz) {
    float d = decbg[tid];
    s_decf[0][tid] = d;
    s_decb[0][tid] = f2b(d);
  } else {
    s_sc[tid] = -__builtin_inff();   // step 0: rows >= 512 never written by phase A
  }
  if (tid == 0) s_mxkey = 0u;        // below key of any real score
  __threadfence();
  __syncthreads();                   // ---- barrier 0 (once) ----

  for (int i = 0; i < Ssz; ++i) {
    const int M   = Ssz + i;         // valid memory rows this step
    const int cur = i & 1;
    const int nxt = cur ^ 1;

    // ---- region X: scores[m] = <mem[m,:], dec_i> via WMMA bf16 (mask folded in) ----
    const int tiles = (M + 15) >> 4;              // >= 32, <= 64
    for (int mt = wave; mt < tiles; mt += 32) {   // wave-uniform: EXEC full for WMMA
      const int r0 = mt << 4;
      const int m  = r0 + (lane & 15);
      const int h  = lane >> 4;
      const unsigned short* arow = memb + (size_t)m * Dsz;
      v8f acc = {0.f, 0.f, 0.f, 0.f, 0.f, 0.f, 0.f, 0.f};
      // unroll 4 (not 16): keeps hoisted B tiles at 32 VGPRs -> no scratch spills
      #pragma unroll 4
      for (int kt = 0; kt < Dsz / 32; ++kt) {
        const int k0 = kt << 5;
        ABPack a, bb;
        // A 16x32 bf16: elems 0..7 -> K = 8h+0..7 ; elems 8..15 -> K = 16+8h+0..7
        a.q[0]  = *(const uint4*)(arow + k0 + 8 * h);
        a.q[1]  = *(const uint4*)(arow + k0 + 16 + 8 * h);
        // B 32x16 bf16: elems 0..7 -> K = 8h+0..7 ; elems 8..15 -> K = 16+8h+0..7; all N cols = dec
        bb.q[0] = *(const uint4*)(&s_decb[cur][k0 + 8 * h]);
        bb.q[1] = *(const uint4*)(&s_decb[cur][k0 + 16 + 8 * h]);
        acc = __builtin_amdgcn_wmma_f32_16x16x32_bf16(false, a.v, false, bb.v,
                                                      (short)0, acc, false, false);
      }
      // C layout: VGPR j = row j (lanes 0-15) / row 8+j (lanes 16-31); take N=0 column
      if (lane == 0) {
        #pragma unroll
        for (int j = 0; j < 8; ++j)
          s_sc[r0 + j] = (r0 + j < M) ? acc[j] : -__builtin_inff();
      } else if (lane == 16) {
        #pragma unroll
        for (int j = 0; j < 8; ++j)
          s_sc[r0 + 8 + j] = (r0 + 8 + j < M) ? acc[j] : -__builtin_inff();
      }
    }
    __syncthreads();                 // ---- barrier A ----

    // ---- region Y: deterministic max via keyed ds_max_u32 ----
    float v = s_sc[tid];
    #pragma unroll
    for (int off = 16; off > 0; off >>= 1) v = fmaxf(v, __shfl_xor(v, off, 32));
    if (lane == 0) atomicMax(&s_mxkey, fkey(v));
    __syncthreads();                 // ---- barrier B ----

    // ---- region Z: exp + per-wave sums ----
    const float mx = funkey(s_mxkey);
    const float e  = expf(s_sc[tid] - mx);   // masked rows: exp(-inf) = 0
    float sv = e;
    #pragma unroll
    for (int off = 16; off > 0; off >>= 1) sv += __shfl_xor(sv, off, 32);
    if (lane == 0) s_redsum[wave] = sv;
    __syncthreads();                 // ---- barrier C ----

    // ---- region W: broadcast-sum partials (fixed order => deterministic) ----
    float sum = 0.f;
    #pragma unroll
    for (int gg = 0; gg < 32; ++gg) sum += s_redsum[gg];
    const float inv  = 1.0f / sum;
    const int   limn = (M + 16 + 15) & ~15;  // next step's tile-covered limit
    s_sc[tid] = (tid >= limn) ? -__builtin_inff() : e * inv; // att (0 for t>=M) / next-step seed
    if (tid == 0) s_mxkey = 0u;              // reset for next step
    __syncthreads();                 // ---- barrier D ----

    // ---- region V: out[d] = sum_m att[m]*mem[m,d] (b128 coalesced), + stage next query ----
    const int dbase = (tid & 63) << 3;       // 8 contiguous d-columns per thread
    const int g     = tid >> 6;              // 16 m-strided groups
    float a0 = 0.f, a1 = 0.f, a2 = 0.f, a3 = 0.f;
    float a4 = 0.f, a5 = 0.f, a6 = 0.f, a7 = 0.f;
    for (int m = g; m < M; m += 16) {
      const float w = s_sc[m];               // wave-uniform -> LDS broadcast
      const uint4 q = *(const uint4*)(memb + (size_t)m * Dsz + dbase);
      a0 = fmaf(w, b2f((unsigned short)(q.x & 0xFFFFu)), a0);
      a1 = fmaf(w, b2f((unsigned short)(q.x >> 16)),     a1);
      a2 = fmaf(w, b2f((unsigned short)(q.y & 0xFFFFu)), a2);
      a3 = fmaf(w, b2f((unsigned short)(q.y >> 16)),     a3);
      a4 = fmaf(w, b2f((unsigned short)(q.z & 0xFFFFu)), a4);
      a5 = fmaf(w, b2f((unsigned short)(q.z >> 16)),     a5);
      a6 = fmaf(w, b2f((unsigned short)(q.w & 0xFFFFu)), a6);
      a7 = fmaf(w, b2f((unsigned short)(q.w >> 16)),     a7);
    }
    s_part[g][dbase + 0] = a0;
    s_part[g][dbase + 1] = a1;
    s_part[g][dbase + 2] = a2;
    s_part[g][dbase + 3] = a3;
    s_part[g][dbase + 4] = a4;
    s_part[g][dbase + 5] = a5;
    s_part[g][dbase + 6] = a6;
    s_part[g][dbase + 7] = a7;
    if (i + 1 < Ssz && tid < Dsz) {          // overlap: stage dec[:, i+1]
      float d = decbg[(size_t)(i + 1) * Dsz + tid];
      s_decf[nxt][tid] = d;
      s_decb[nxt][tid] = f2b(d);
    }
    __syncthreads();                 // ---- barrier E ----

    // ---- region U: combine, residual, write output + append memory row ----
    if (tid < Dsz) {
      float sum2 = s_decf[cur][tid];
      #pragma unroll
      for (int gg = 0; gg < 16; ++gg) sum2 += s_part[gg][tid];
      outb[(size_t)i * Dsz + tid] = sum2;                    // f32 output
      memb[(size_t)(Ssz + i) * Dsz + tid] = f2b(sum2);       // bf16 memory append
    }
    __threadfence();   // make appended row visible to next step's global loads
    __syncthreads();                 // ---- barrier F ----
  }
}

extern "C" void kernel_launch(void* const* d_in, const int* in_sizes, int n_in,
                              void* d_out, int out_size, void* d_ws, size_t ws_size,
                              hipStream_t stream) {
  const float* enc = (const float*)d_in[0];   // [32, 512, 512] f32
  const float* dec = (const float*)d_in[1];   // [32, 512, 512] f32
  float* out = (float*)d_out;                 // [32, 512, 512] f32
  unsigned short* memb = (unsigned short*)d_ws; // 32 * 1024 * 512 bf16 = 32 MB
  (void)in_sizes; (void)n_in; (void)out_size; (void)ws_size;
  att_incremental_kernel<<<dim3(Bsz), dim3(1024), 0, stream>>>(enc, dec, out, memb);
}
